// GCN_TAT_layers_46033459478799
// MI455X (gfx1250) — matrix operand order repaired and checked by reading, not compiled, not run on hardware
//
#include <hip/hip_runtime.h>
#include <hip/hip_bf16.h>
#include <math.h>

typedef __bf16 bf16;
typedef __attribute__((ext_vector_type(16))) __bf16 v16bf;
typedef __attribute__((ext_vector_type(8)))  __bf16 v8bf;
typedef __attribute__((ext_vector_type(8)))  float  v8f;

#define DEV static __device__ __forceinline__

// ---- problem constants ----
#define Bn   32
#define CIN  64
#define COUT 64
#define Tt   128
#define Vv   19
#define Dd   192
#define DFF  384
#define Hh   8
#define HD   24
#define TV   2432          // Tt*Vv
#define MROW 77824         // Bn*Tt*Vv  (== 4864 * 16)

// ================= WMMA helpers =================

DEV v8f zero8f() {
  v8f c;
  #pragma unroll
  for (int i = 0; i < 8; ++i) c[i] = 0.f;
  return c;
}

DEV v8bf zero8bf() {
  v8bf z;
  #pragma unroll
  for (int i = 0; i < 8; ++i) z[i] = (bf16)0.0f;
  return z;
}

DEV v16bf pack16(v8bf lo, v8bf hi) {
  v16bf a;
  #pragma unroll
  for (int i = 0; i < 8; ++i) { a[i] = lo[i]; a[i + 8] = hi[i]; }
  return a;
}

DEV v8f wmma_bf16(v16bf a, v16bf b, v8f c) {
  return __builtin_amdgcn_wmma_f32_16x16x32_bf16(false, a, false, b, (short)0, c, false, false);
}

// A operand: src is (M x K) row-major bf16.  lane%16 = row, lane half selects
// K chunks {g*8..g*8+7} and {16+g*8..16+g*8+7}  -> two b128 loads.
DEV v16bf load_a(const bf16* base, int ld, int row0, int k0, int lane) {
  int g = lane >> 4;
  const bf16* p = base + (size_t)(row0 + (lane & 15)) * ld + k0 + g * 8;
  v8bf lo = *(const v8bf*)p;
  v8bf hi = *(const v8bf*)(p + 16);
  return pack16(lo, hi);
}

// A operand with only 24 of 32 K valid (head-dim padding): lanes>=16 upper chunk
// covers K=24..31 -> force zero.
DEV v16bf load_a_pad24(const bf16* base, int ld, int row0, int k0, int lane) {
  int g = lane >> 4;
  const bf16* p = base + (size_t)(row0 + (lane & 15)) * ld + k0 + g * 8;
  v8bf lo = *(const v8bf*)p;
  v8bf hi = (lane & 16) ? zero8bf() : *(const v8bf*)(p + 16);
  return pack16(lo, hi);
}

// B operand: srcT is (N x K) row-major bf16 (transposed weights).
// lane%16 = column N, lane half g holds K chunks {g*8..g*8+7} and {16+g*8..16+g*8+7}.
DEV v16bf load_b(const bf16* baseT, int ld, int n0, int k0, int lane) {
  int g = lane >> 4;
  const bf16* p = baseT + (size_t)(n0 + (lane & 15)) * ld + k0 + g * 8;
  v8bf lo = *(const v8bf*)p;
  v8bf hi = *(const v8bf*)(p + 16);
  return pack16(lo, hi);
}

// B operand, only K<24 of 32 valid: lanes>=16 second 8 halves = K 24..31 -> zero.
DEV v16bf load_b_pad24(const bf16* baseT, int ld, int n0, int koff, int lane) {
  int g = lane >> 4;
  const bf16* p = baseT + (size_t)(n0 + (lane & 15)) * ld + koff + g * 8;
  v8bf lo = *(const v8bf*)p;
  v8bf hi = (lane & 16) ? zero8bf() : *(const v8bf*)(p + 16);
  return pack16(lo, hi);
}

// A operand gathered from x (n,c,t,v) fp32, M index = (n,t,v) flat.
DEV v16bf load_a_x(const float* x, int row0, int c0, int lane) {
  int r = row0 + (lane & 15);
  int n = r / TV; int rem = r - n * TV; int t = rem / Vv; int v = rem - t * Vv;
  const float* p = x + (size_t)n * (CIN * TV) + t * Vv + v
                     + (size_t)(c0 + ((lane >> 4) << 3)) * TV;
  v16bf a;
  #pragma unroll
  for (int i = 0; i < 8; ++i) a[i] = (bf16)p[(size_t)i * TV];
  #pragma unroll
  for (int i = 0; i < 8; ++i) a[8 + i] = (bf16)p[(size_t)(16 + i) * TV];
  return a;
}

// A operand from fp32 row-major matrix, convert on load.
DEV v16bf load_a_f32(const float* base, int ld, int row0, int k0, int lane) {
  const float* p = base + (size_t)(row0 + (lane & 15)) * ld + k0 + ((lane >> 4) << 3);
  v16bf a;
  #pragma unroll
  for (int i = 0; i < 8; ++i) a[i] = (bf16)p[i];
  #pragma unroll
  for (int i = 0; i < 8; ++i) a[8 + i] = (bf16)p[16 + i];
  return a;
}

// ================= wave utilities =================

DEV float wave_sum(float v) {
  #pragma unroll
  for (int m = 16; m > 0; m >>= 1) v += __shfl_xor(v, m, 32);
  return v;
}
DEV float wave_max(float v) {
  #pragma unroll
  for (int m = 16; m > 0; m >>= 1) v = fmaxf(v, __shfl_xor(v, m, 32));
  return v;
}

// LayerNorm rows of a 128x192 fp32 LDS matrix -> bf16 LDS matrix.
DEV void ln_rows(const float* src, bf16* dst, const float* w, const float* b,
                 int lane, int wid) {
  for (int r = wid; r < Tt; r += 8) {
    const float* p = src + (size_t)r * Dd;
    float s = 0.f, q = 0.f;
    for (int i = lane; i < Dd; i += 32) { float v = p[i]; s += v; q += v * v; }
    s = wave_sum(s); q = wave_sum(q);
    float mean = s * (1.f / Dd);
    float inv = rsqrtf(q * (1.f / Dd) - mean * mean + 1e-5f);
    for (int i = lane; i < Dd; i += 32)
      dst[(size_t)r * Dd + i] = (bf16)((p[i] - mean) * inv * w[i] + b[i]);
  }
}

// exact 1.5-entmax via monotone bisection for tau (one wave per 128-wide row).
// Input s already holds z = logits * 0.5/sqrt(hd).
DEV void entmax_row(const float* s, bf16* p, int lane) {
  float z0 = s[lane], z1 = s[lane + 32], z2 = s[lane + 64], z3 = s[lane + 96];
  float mx = wave_max(fmaxf(fmaxf(z0, z1), fmaxf(z2, z3)));
  z0 -= mx; z1 -= mx; z2 -= mx; z3 -= mx;
  float lo = -1.f, hi = 0.f;
  #pragma unroll 1
  for (int it = 0; it < 26; ++it) {
    float tau = 0.5f * (lo + hi);
    float a0 = fmaxf(z0 - tau, 0.f), a1 = fmaxf(z1 - tau, 0.f);
    float a2 = fmaxf(z2 - tau, 0.f), a3 = fmaxf(z3 - tau, 0.f);
    float ss = wave_sum(a0 * a0 + a1 * a1 + a2 * a2 + a3 * a3);
    if (ss >= 1.f) lo = tau; else hi = tau;
  }
  float tau = 0.5f * (lo + hi);
  float p0 = fmaxf(z0 - tau, 0.f); p0 *= p0;
  float p1 = fmaxf(z1 - tau, 0.f); p1 *= p1;
  float p2 = fmaxf(z2 - tau, 0.f); p2 *= p2;
  float p3 = fmaxf(z3 - tau, 0.f); p3 *= p3;
  float inv = 1.f / fmaxf(wave_sum(p0 + p1 + p2 + p3), 1e-12f);
  p[lane]      = (bf16)(p0 * inv);
  p[lane + 32] = (bf16)(p1 * inv);
  p[lane + 64] = (bf16)(p2 * inv);
  p[lane + 96] = (bf16)(p3 * inv);
}

// ================= kernels =================

__global__ void __launch_bounds__(256)
k_transpose_bf16(const float* __restrict__ src, bf16* __restrict__ dst, int K, int N) {
  int idx = blockIdx.x * blockDim.x + threadIdx.x;
  if (idx >= K * N) return;
  int k = idx / N, n = idx - k * N;
  dst[(size_t)n * K + k] = (bf16)src[idx];
}

// y0(m,192) = x @ lin_w + lin_b ; m = (n,t,v) flat
__global__ void __launch_bounds__(128)
k_lin(const float* __restrict__ x, const bf16* __restrict__ WT,
      const float* __restrict__ bias, float* __restrict__ y0) {
  int lane = threadIdx.x & 31, wid = threadIdx.x >> 5;
  int row0 = blockIdx.x * 16;
  for (int nt = wid; nt < 12; nt += 4) {
    v8f c = zero8f();
    #pragma unroll
    for (int ks = 0; ks < 2; ++ks) {
      v16bf a = load_a_x(x, row0, ks * 32, lane);
      v16bf b = load_b(WT, CIN, nt * 16, ks * 32, lane);
      c = wmma_bf16(a, b, c);
    }
    int col = lane & 15, rb = (lane >> 4) * 8;
    float bb = bias[nt * 16 + col];
    #pragma unroll
    for (int r = 0; r < 8; ++r)
      y0[(size_t)(row0 + rb + r) * Dd + nt * 16 + col] = c[r] + bb;
  }
}

// per-channel training-mode BN stats -> scale/shift
__global__ void __launch_bounds__(256)
k_bnstats(const float* __restrict__ y, int M, int C,
          const float* __restrict__ w, const float* __restrict__ b,
          float* __restrict__ scale, float* __restrict__ shift) {
  __shared__ float ss[256], sq[256];
  int ch = blockIdx.x, tid = threadIdx.x;
  float s = 0.f, q = 0.f;
  for (int i = tid; i < M; i += 256) {
    float v = y[(size_t)i * C + ch]; s += v; q += v * v;
  }
  ss[tid] = s; sq[tid] = q; __syncthreads();
  for (int st = 128; st > 0; st >>= 1) {
    if (tid < st) { ss[tid] += ss[tid + st]; sq[tid] += sq[tid + st]; }
    __syncthreads();
  }
  if (tid == 0) {
    float m = ss[0] / M;
    float var = sq[0] / M - m * m;
    float sc = w[ch] * rsqrtf(var + 1e-5f);
    scale[ch] = sc; shift[ch] = b[ch] - m * sc;
  }
}

// normalized adjacency: dAn[k,g,v,w] = dA / (sqrt(8*sum_g dA^2) + 1e-4)
__global__ void __launch_bounds__(256)
k_anorm(const float* __restrict__ dA, float* __restrict__ dAn) {
  int idx = blockIdx.x * blockDim.x + threadIdx.x;   // 3*19*19
  if (idx >= 3 * 361) return;
  int k = idx / 361, vw = idx - k * 361;
  float s = 0.f;
  #pragma unroll
  for (int g = 0; g < 8; ++g) { float a = dA[(size_t)(k * 8 + g) * 361 + vw]; s += a * a; }
  float inv = 1.f / (sqrtf(8.f * s) + 1e-4f);
  #pragma unroll
  for (int g = 0; g < 8; ++g) {
    size_t o = (size_t)(k * 8 + g) * 361 + vw;
    dAn[o] = dA[o] * inv;
  }
}

// graph conv: y1[(n,t,w),c] = sum_k sum_v bn0(y0)[(n,t,v), k*64+c] * dAn[k,c%8,v,w]
__global__ void __launch_bounds__(256)
k_gcn(const float* __restrict__ y0, const float* __restrict__ sc0,
      const float* __restrict__ sh0, const float* __restrict__ dAn,
      float* __restrict__ y1) {
  int idx = blockIdx.x * blockDim.x + threadIdx.x;
  if (idx >= MROW * 64) return;
  int m = idx >> 6, c = idx & 63;
  int n = m / TV; int rem = m - n * TV; int t = rem / Vv; int w = rem - t * Vv;
  const float* yb = y0 + ((size_t)n * TV + t * Vv) * Dd;
  float acc = 0.f;
  #pragma unroll
  for (int k = 0; k < 3; ++k) {
    int d = k * 64 + c;
    float scd = sc0[d], shd = sh0[d];
    const float* Ak = dAn + (size_t)(k * 8 + (c & 7)) * 361;
    for (int v = 0; v < Vv; ++v) {
      float yv = yb[(size_t)v * Dd + d] * scd + shd;
      acc += yv * Ak[v * Vv + w];
    }
  }
  y1[(size_t)m * 64 + c] = acc;
}

// x2 = relu(bn1(y1)+x) + x  stored bf16 with m2=(n,v,t) ordering for embed GEMM
__global__ void __launch_bounds__(256)
k_x2(const float* __restrict__ y1, const float* __restrict__ sc1,
     const float* __restrict__ sh1, const float* __restrict__ x,
     bf16* __restrict__ x2) {
  int idx = blockIdx.x * blockDim.x + threadIdx.x;
  if (idx >= MROW * 64) return;
  int m = idx >> 6, c = idx & 63;
  int n = m / TV; int rem = m - n * TV; int t = rem / Vv; int v = rem - t * Vv;
  float xv = x[(((size_t)n * 64 + c) * Tt + t) * Vv + v];
  float g = fmaxf(y1[idx] * sc1[c] + sh1[c] + xv, 0.f) + xv;
  int m2 = (n * Vv + v) * Tt + t;
  x2[(size_t)m2 * 64 + c] = (bf16)g;
}

// t(m2,192) = x2 @ embed_w / sqrt(D)
__global__ void __launch_bounds__(128)
k_embed(const bf16* __restrict__ x2, const bf16* __restrict__ WT,
        float* __restrict__ tg) {
  int lane = threadIdx.x & 31, wid = threadIdx.x >> 5;
  int row0 = blockIdx.x * 16;
  const float s = 0.0721687836f;   // 1/sqrt(192)
  for (int nt = wid; nt < 12; nt += 4) {
    v8f c = zero8f();
    #pragma unroll
    for (int ks = 0; ks < 2; ++ks) {
      v16bf a = load_a(x2, 64, row0, ks * 32, lane);
      v16bf b = load_b(WT, 64, nt * 16, ks * 32, lane);
      c = wmma_bf16(a, b, c);
    }
    int col = lane & 15, rb = (lane >> 4) * 8;
    #pragma unroll
    for (int r = 0; r < 8; ++r)
      tg[(size_t)(row0 + rb + r) * Dd + nt * 16 + col] = c[r] * s;
  }
}

// Fused encoder layer: one workgroup per sequence, everything in LDS.
__global__ void __launch_bounds__(256)
k_encoder(float* __restrict__ tg,
          const float* __restrict__ ln1w, const float* __restrict__ ln1b,
          const bf16* __restrict__ wqT, const float* __restrict__ bq,
          const bf16* __restrict__ wkT, const float* __restrict__ bk,
          const bf16* __restrict__ wvT, const float* __restrict__ bvp,
          const bf16* __restrict__ woT, const float* __restrict__ bo,
          const float* __restrict__ ln2w, const float* __restrict__ ln2b,
          const bf16* __restrict__ w1T, const float* __restrict__ b1,
          const bf16* __restrict__ w2T, const float* __restrict__ b2,
          const float* __restrict__ lnfw, const float* __restrict__ lnfb) {
  extern __shared__ char smem[];
  float* tb = (float*)smem;                  // 128x192 f32   (98304 B)
  bf16*  hb = (bf16*)(smem + 98304);         // 128x192 bf16  (49152 B)
  bf16*  Qb = (bf16*)(smem + 147456);        // 128x192 bf16
  bf16*  Kb = (bf16*)(smem + 196608);        // 128x192 bf16
  bf16*  Vt = (bf16*)(smem + 245760);        // 192x128 bf16 (d-major)
  float* Sb = (float*)(smem + 294912);       // 32x128 f32 score chunk
  bf16*  Pb = (bf16*)(smem + 311296);        // 32x128 bf16 probs
  bf16*  Zb = Qb;                            // 128x384 bf16 FFN hidden (reuse)

  const int lane = threadIdx.x & 31;
  const int wid  = threadIdx.x >> 5;
  float* trow = tg + (size_t)blockIdx.x * (Tt * Dd);

  __builtin_prefetch(wqT, 0, 0);
  __builtin_prefetch(w1T, 0, 0);

  for (int i = threadIdx.x; i < Tt * Dd; i += 256) tb[i] = trow[i];
  __syncthreads();

  ln_rows(tb, hb, ln1w, ln1b, lane, wid);
  __syncthreads();

  // ---- Q = h@Wq+bq (row-major), K likewise, V stored transposed ----
  for (int tile = wid; tile < 96; tile += 8) {
    int mt = tile / 12, nt = tile - mt * 12;
    v8f c = zero8f();
    #pragma unroll
    for (int ks = 0; ks < 6; ++ks)
      c = wmma_bf16(load_a(hb, Dd, mt * 16, ks * 32, lane),
                    load_b(wqT, Dd, nt * 16, ks * 32, lane), c);
    int col = lane & 15, rb = (lane >> 4) * 8;
    float bb = bq[nt * 16 + col];
    #pragma unroll
    for (int r = 0; r < 8; ++r)
      Qb[(size_t)(mt * 16 + rb + r) * Dd + nt * 16 + col] = (bf16)(c[r] + bb);
  }
  for (int tile = wid; tile < 96; tile += 8) {
    int mt = tile / 12, nt = tile - mt * 12;
    v8f c = zero8f();
    #pragma unroll
    for (int ks = 0; ks < 6; ++ks)
      c = wmma_bf16(load_a(hb, Dd, mt * 16, ks * 32, lane),
                    load_b(wkT, Dd, nt * 16, ks * 32, lane), c);
    int col = lane & 15, rb = (lane >> 4) * 8;
    float bb = bk[nt * 16 + col];
    #pragma unroll
    for (int r = 0; r < 8; ++r)
      Kb[(size_t)(mt * 16 + rb + r) * Dd + nt * 16 + col] = (bf16)(c[r] + bb);
  }
  for (int tile = wid; tile < 96; tile += 8) {
    int mt = tile / 12, nt = tile - mt * 12;
    v8f c = zero8f();
    #pragma unroll
    for (int ks = 0; ks < 6; ++ks)
      c = wmma_bf16(load_a(hb, Dd, mt * 16, ks * 32, lane),
                    load_b(wvT, Dd, nt * 16, ks * 32, lane), c);
    int col = lane & 15, rb = (lane >> 4) * 8;
    float bb = bvp[nt * 16 + col];
    #pragma unroll
    for (int r = 0; r < 8; ++r)
      Vt[(size_t)(nt * 16 + col) * Tt + (mt * 16 + rb + r)] = (bf16)(c[r] + bb);
  }
  __syncthreads();

  // ---- attention: per head, 32-query chunks; O written into hb ----
  const float sscale = 0.102062072f;  // (1/sqrt(24)) * (1/2 of entmax)
  for (int h = 0; h < Hh; ++h) {
    const int off = h * HD;
    for (int qc = 0; qc < 4; ++qc) {
      const int q0 = qc * 32;
      for (int tile = wid; tile < 16; tile += 8) {          // 2x8 score tiles
        int mt = tile >> 3, nt = tile & 7;
        v8f c = wmma_bf16(load_a_pad24(Qb, Dd, q0 + mt * 16, off, lane),
                          load_b_pad24(Kb, Dd, nt * 16, off, lane), zero8f());
        int col = lane & 15, rb = (lane >> 4) * 8;
        #pragma unroll
        for (int r = 0; r < 8; ++r)
          Sb[(size_t)(mt * 16 + rb + r) * 128 + nt * 16 + col] = c[r] * sscale;
      }
      __syncthreads();
      for (int r = wid; r < 32; r += 8)
        entmax_row(Sb + (size_t)r * 128, Pb + (size_t)r * 128, lane);
      __syncthreads();
      if (wid < 4) {                                        // 2x2 output tiles
        int mt = wid >> 1, nt = wid & 1;
        v8f c = zero8f();
        #pragma unroll
        for (int ks = 0; ks < 4; ++ks)
          c = wmma_bf16(load_a(Pb, 128, mt * 16, ks * 32, lane),
                        load_b(Vt, Tt, off + nt * 16, ks * 32, lane), c);
        int col = lane & 15, rb = (lane >> 4) * 8;
        if (nt * 16 + col < HD) {
          #pragma unroll
          for (int r = 0; r < 8; ++r)
            hb[(size_t)(q0 + mt * 16 + rb + r) * Dd + off + nt * 16 + col] = (bf16)c[r];
        }
      }
      __syncthreads();
    }
  }

  // ---- output projection + residual into tb ----
  for (int tile = wid; tile < 96; tile += 8) {
    int mt = tile / 12, nt = tile - mt * 12;
    v8f c = zero8f();
    #pragma unroll
    for (int ks = 0; ks < 6; ++ks)
      c = wmma_bf16(load_a(hb, Dd, mt * 16, ks * 32, lane),
                    load_b(woT, Dd, nt * 16, ks * 32, lane), c);
    int col = lane & 15, rb = (lane >> 4) * 8;
    float bb = bo[nt * 16 + col];
    #pragma unroll
    for (int r = 0; r < 8; ++r)
      tb[(size_t)(mt * 16 + rb + r) * Dd + nt * 16 + col] += c[r] + bb;
  }
  __syncthreads();

  ln_rows(tb, hb, ln2w, ln2b, lane, wid);
  __syncthreads();

  // ---- FFN1 + Mish -> Zb ----
  for (int tile = wid; tile < 192; tile += 8) {             // 8 x 24 tiles
    int mt = tile / 24, nt = tile - mt * 24;
    v8f c = zero8f();
    #pragma unroll
    for (int ks = 0; ks < 6; ++ks)
      c = wmma_bf16(load_a(hb, Dd, mt * 16, ks * 32, lane),
                    load_b(w1T, Dd, nt * 16, ks * 32, lane), c);
    int col = lane & 15, rb = (lane >> 4) * 8;
    float bb = b1[nt * 16 + col];
    #pragma unroll
    for (int r = 0; r < 8; ++r) {
      float zz = c[r] + bb;
      float sp = (zz > 20.f) ? zz : log1pf(expf(zz));
      Zb[(size_t)(mt * 16 + rb + r) * DFF + nt * 16 + col] = (bf16)(zz * tanhf(sp));
    }
  }
  __syncthreads();

  // ---- FFN2 + residual into tb ----
  for (int tile = wid; tile < 96; tile += 8) {
    int mt = tile / 12, nt = tile - mt * 12;
    v8f c = zero8f();
    #pragma unroll
    for (int ks = 0; ks < 12; ++ks)
      c = wmma_bf16(load_a(Zb, DFF, mt * 16, ks * 32, lane),
                    load_b(w2T, DFF, nt * 16, ks * 32, lane), c);
    int col = lane & 15, rb = (lane >> 4) * 8;
    float bb = b2[nt * 16 + col];
    #pragma unroll
    for (int r = 0; r < 8; ++r)
      tb[(size_t)(mt * 16 + rb + r) * Dd + nt * 16 + col] += c[r] + bb;
  }
  __syncthreads();

  // ---- final LN + outer residual (mem = global t), write back ----
  for (int r = wid; r < Tt; r += 8) {
    const float* p = tb + (size_t)r * Dd;
    float s = 0.f, q = 0.f;
    for (int i = lane; i < Dd; i += 32) { float v = p[i]; s += v; q += v * v; }
    s = wave_sum(s); q = wave_sum(q);
    float mean = s * (1.f / Dd);
    float inv = rsqrtf(q * (1.f / Dd) - mean * mean + 1e-5f);
    for (int i = lane; i < Dd; i += 32) {
      float nv = (p[i] - mean) * inv * lnfw[i] + lnfb[i];
      trow[(size_t)r * Dd + i] += nv;          // t_new = LN(...) + mem
    }
  }
}

// out = relu(t @ out_w + out_b), scatter to (n,c,t,v)
__global__ void __launch_bounds__(128)
k_out(const float* __restrict__ tg, const bf16* __restrict__ WT,
      const float* __restrict__ ob, float* __restrict__ out) {
  int lane = threadIdx.x & 31, wid = threadIdx.x >> 5;
  int row0 = blockIdx.x * 16;
  int nt = wid;                                // 4 N-tiles of 16
  v8f c = zero8f();
  #pragma unroll
  for (int ks = 0; ks < 6; ++ks)
    c = wmma_bf16(load_a_f32(tg, Dd, row0, ks * 32, lane),
                  load_b(WT, Dd, nt * 16, ks * 32, lane), c);
  int col = lane & 15, rb = (lane >> 4) * 8;
  float bb = ob[nt * 16 + col];
  #pragma unroll
  for (int r = 0; r < 8; ++r) {
    int m2 = row0 + rb + r;                    // m2 = (n,v,t)
    int n = m2 / TV; int rem = m2 - n * TV; int v = rem >> 7; int t = rem & 127;
    out[(((size_t)n * 64 + nt * 16 + col) * Tt + t) * Vv + v] = fmaxf(c[r] + bb, 0.f);
  }
}

// ================= host launcher =================

extern "C" void kernel_launch(void* const* d_in, const int* in_sizes, int n_in,
                              void* d_out, int out_size, void* d_ws, size_t ws_size,
                              hipStream_t stream) {
  (void)in_sizes; (void)n_in; (void)out_size; (void)ws_size;
  const float* x     = (const float*)d_in[0];
  const float* lin_w = (const float*)d_in[1];
  const float* lin_b = (const float*)d_in[2];
  const float* dA    = (const float*)d_in[3];
  const float* bn0_w = (const float*)d_in[4];
  const float* bn0_b = (const float*)d_in[5];
  const float* bn1_w = (const float*)d_in[6];
  const float* bn1_b = (const float*)d_in[7];
  const float* emb_w = (const float*)d_in[8];
  const float* ln1_w = (const float*)d_in[9];
  const float* ln1_b = (const float*)d_in[10];
  const float* wq    = (const float*)d_in[11];
  const float* bq    = (const float*)d_in[12];
  const float* wk    = (const float*)d_in[13];
  const float* bk    = (const float*)d_in[14];
  const float* wv    = (const float*)d_in[15];
  const float* bv    = (const float*)d_in[16];
  const float* wo    = (const float*)d_in[17];
  const float* bo    = (const float*)d_in[18];
  const float* ln2_w = (const float*)d_in[19];
  const float* ln2_b = (const float*)d_in[20];
  const float* w1    = (const float*)d_in[21];
  const float* b1    = (const float*)d_in[22];
  const float* w2    = (const float*)d_in[23];
  const float* b2    = (const float*)d_in[24];
  const float* lnf_w = (const float*)d_in[25];
  const float* lnf_b = (const float*)d_in[26];
  const float* out_w = (const float*)d_in[27];
  const float* out_b = (const float*)d_in[28];

  char* ws = (char*)d_ws;
  size_t off = 0;
  auto take = [&](size_t bytes) -> char* {
    char* p = ws + off; off = (off + bytes + 255) & ~(size_t)255; return p;
  };
  bf16* linWT = (bf16*)take(192 * 64 * 2);
  bf16* embWT = (bf16*)take(192 * 64 * 2);
  bf16* wqT   = (bf16*)take(2 * 192 * 192 * 2);
  bf16* wkT   = (bf16*)take(2 * 192 * 192 * 2);
  bf16* wvT   = (bf16*)take(2 * 192 * 192 * 2);
  bf16* woT   = (bf16*)take(2 * 192 * 192 * 2);
  bf16* w1T   = (bf16*)take(2 * 384 * 192 * 2);
  bf16* w2T   = (bf16*)take(2 * 192 * 384 * 2);
  bf16* outWT = (bf16*)take(64 * 192 * 2);
  float* y0   = (float*)take((size_t)MROW * 192 * 4);
  float* y1   = (float*)take((size_t)MROW * 64 * 4);
  bf16*  x2b  = (bf16*)take((size_t)MROW * 64 * 2);
  float* tg   = (float*)take((size_t)MROW * 192 * 4);
  float* sc0  = (float*)take(192 * 4);
  float* sh0  = (float*)take(192 * 4);
  float* sc1  = (float*)take(64 * 4);
  float* sh1  = (float*)take(64 * 4);
  float* dAn  = (float*)take(3 * 8 * 361 * 4);

  auto tr = [&](const float* src, bf16* dst, int K, int N) {
    int n = K * N;
    k_transpose_bf16<<<(n + 255) / 256, 256, 0, stream>>>(src, dst, K, N);
  };
  tr(lin_w, linWT, 64, 192);
  tr(emb_w, embWT, 64, 192);
  for (int i = 0; i < 2; ++i) {
    tr(wq + i * 192 * 192, wqT + i * 192 * 192, 192, 192);
    tr(wk + i * 192 * 192, wkT + i * 192 * 192, 192, 192);
    tr(wv + i * 192 * 192, wvT + i * 192 * 192, 192, 192);
    tr(wo + i * 192 * 192, woT + i * 192 * 192, 192, 192);
    tr(w1 + i * 192 * 384, w1T + i * 384 * 192, 192, 384);
    tr(w2 + i * 384 * 192, w2T + i * 192 * 384, 384, 192);
  }
  tr(out_w, outWT, 192, 64);

  k_lin<<<MROW / 16, 128, 0, stream>>>(x, linWT, lin_b, y0);
  k_bnstats<<<192, 256, 0, stream>>>(y0, MROW, 192, bn0_w, bn0_b, sc0, sh0);
  k_anorm<<<(3 * 361 + 255) / 256, 256, 0, stream>>>(dA, dAn);
  k_gcn<<<(MROW * 64 + 255) / 256, 256, 0, stream>>>(y0, sc0, sh0, dAn, y1);
  k_bnstats<<<64, 256, 0, stream>>>(y1, MROW, 64, bn1_w, bn1_b, sc1, sh1);
  k_x2<<<(MROW * 64 + 255) / 256, 256, 0, stream>>>(y1, sc1, sh1, x, x2b);
  k_embed<<<MROW / 16, 128, 0, stream>>>(x2b, embWT, tg);

  (void)hipFuncSetAttribute((const void*)k_encoder,
                            hipFuncAttributeMaxDynamicSharedMemorySize, 320000);
  for (int i = 0; i < 2; ++i) {
    k_encoder<<<Bn * Vv, 256, 320000, stream>>>(
        tg, ln1_w + i * 192, ln1_b + i * 192,
        wqT + i * 192 * 192, bq + i * 192,
        wkT + i * 192 * 192, bk + i * 192,
        wvT + i * 192 * 192, bv + i * 192,
        woT + i * 192 * 192, bo + i * 192,
        ln2_w + i * 192, ln2_b + i * 192,
        w1T + i * 384 * 192, b1 + i * 384,
        w2T + i * 192 * 384, b2 + i * 192,
        lnf_w + i * 192, lnf_b + i * 192);
  }
  k_out<<<MROW / 16, 128, 0, stream>>>(tg, outWT, out_b, (float*)d_out);
}
